// LlamaSdpaAttention_17119739641863
// MI455X (gfx1250) — compile-verified, matrix operand order checked by reading, not run on hardware
//
#include <hip/hip_runtime.h>

// ---------------------------------------------------------------------------
// Problem constants (match reference: B=2, S=2048, H=32, KV=8, D=128)
// ---------------------------------------------------------------------------
#define BATCH   2
#define SEQ     2048
#define NTOK    (BATCH * SEQ)     // 4096
#define HEADS   32
#define KVH     8
#define HDIM    128
#define HID     (HEADS * HDIM)    // 4096
#define KVD     (KVH * HDIM)      // 1024

typedef __attribute__((ext_vector_type(16))) __bf16 v16bf;
typedef __attribute__((ext_vector_type(8)))  __bf16 v8bf;
typedef __attribute__((ext_vector_type(8)))  float  v8f;
typedef __attribute__((ext_vector_type(4)))  unsigned int u32x4;
typedef __attribute__((ext_vector_type(8)))  int  i32x8;
typedef __attribute__((ext_vector_type(4)))  int  i32x4;

#define NEG_BIG (-3.0e38f)

// ---- Tensor Data Mover availability / arity detection ---------------------
#if defined(__has_builtin)
#  if __has_builtin(__builtin_amdgcn_tensor_load_to_lds) && \
      __has_builtin(__builtin_amdgcn_s_wait_tensorcnt)
#    define HAVE_TDM 1
#  endif
#endif
#if __has_include(<hip/amd_detail/amd_gfx1250_TDM.h>)
#  define TDM_6ARG 1     // therock-10.0 headers => 6-arg builtin form
#endif

// ---------------------------------------------------------------------------
// Fragment loader for V_WMMA_F32_16X16X32_BF16.
// 16-bit A (16x32) layout (ISA 7.12.2): lanes 0-15 hold row M=lane,
// K = {0..7, 16..23}; lanes 16-31 hold row M=lane-16, K = {8..15, 24..31}.
// B (32x16) mirrors this with lane indexing N when the tile is stored
// N-major (K contiguous per row) — so both A and B fragments are two
// contiguous 16-byte loads per lane from a row of 32 bf16.
// rstride must be a multiple of 8 elements (16B) for aligned v8bf loads.
// ---------------------------------------------------------------------------
__device__ __forceinline__ v16bf ld_frag(const __bf16* base, int rstride) {
    const int lane = threadIdx.x & 31;
    const int r    = lane & 15;
    const int hi   = (lane >> 4) << 3;      // 0 or 8
    const __bf16* p = base + r * rstride + hi;
    v8bf lo = *reinterpret_cast<const v8bf*>(p);
    v8bf up = *reinterpret_cast<const v8bf*>(p + 16);
    v16bf f;
#pragma unroll
    for (int i = 0; i < 8; ++i) { f[i] = lo[i]; f[i + 8] = up[i]; }
    return f;
}

__device__ __forceinline__ v8f wmma_bf16(v16bf a, v16bf b, v8f c) {
    return __builtin_amdgcn_wmma_f32_16x16x32_bf16(
        /*neg_a=*/false, a, /*neg_b=*/false, b,
        /*c_mod=*/(short)0, c, /*reuse_a=*/false, /*reuse_b=*/false);
}

// ---------------------------------------------------------------------------
// Staging helpers (convert fp32 -> bf16 on the fly; bf16 passthrough)
// ---------------------------------------------------------------------------
__device__ __forceinline__ void load4_cvt(const float* p, __bf16* d) {
    const float4 v = *reinterpret_cast<const float4*>(p);
    d[0] = (__bf16)v.x; d[1] = (__bf16)v.y; d[2] = (__bf16)v.z; d[3] = (__bf16)v.w;
}
__device__ __forceinline__ void load4_cvt(const __bf16* p, __bf16* d) {
    d[0] = p[0]; d[1] = p[1]; d[2] = p[2]; d[3] = p[3];
}
__device__ __forceinline__ void store_out(float*  p, float v) { *p = v; }
__device__ __forceinline__ void store_out(__bf16* p, float v) { *p = (__bf16)v; }

// ---------------------------------------------------------------------------
// GEMM: C[M,N] = A[M,K] * B[K,N].  A is fp32 or bf16, B is fp32 (converted to
// bf16 while staging), C is bf16 or fp32.  Macro tile 128x128, 256 threads =
// 8 waves, each wave computes a 32x64 region = 2x4 WMMA 16x16 tiles.
// BK = 32 == one WMMA K-step.
// ---------------------------------------------------------------------------
#define GBM 128
#define GBN 128
#define GBK 32
#define GLD 40      // padded LDS row stride (bf16 elems), 80B = 16B aligned

template <typename TA, typename TC>
__global__ __launch_bounds__(256) void gemm_wmma(
    const TA* __restrict__ A, const float* __restrict__ B, TC* __restrict__ C,
    int M, int N, int K)
{
    __shared__ __bf16 As[GBM * GLD];
    __shared__ __bf16 Bt[GBN * GLD];     // transposed: [n][k]

    const int tid  = threadIdx.x;
    const int row0 = blockIdx.y * GBM;
    const int n0   = blockIdx.x * GBN;
    const int wid  = tid >> 5;
    const int wm   = (wid & 3) * 32;     // wave row offset within macro tile
    const int wn   = (wid >> 2) * 64;    // wave col offset within macro tile

    v8f acc[2][4];
#pragma unroll
    for (int i = 0; i < 2; ++i)
#pragma unroll
        for (int j = 0; j < 4; ++j) acc[i][j] = {};

    for (int k0 = 0; k0 < K; k0 += GBK) {
        // ---- stage A tile (128 x 32) as bf16, row-major ----
#pragma unroll
        for (int i = 0; i < 4; ++i) {
            int e = tid * 4 + i * 1024;
            int r = e >> 5, c = e & 31;
            __bf16 tmp[4];
            load4_cvt(&A[(size_t)(row0 + r) * K + k0 + c], tmp);
            As[r * GLD + c + 0] = tmp[0]; As[r * GLD + c + 1] = tmp[1];
            As[r * GLD + c + 2] = tmp[2]; As[r * GLD + c + 3] = tmp[3];
        }
        // ---- stage B tile (32 x 128) transposed into Bt[n][k] ----
#pragma unroll
        for (int i = 0; i < 4; ++i) {
            int e  = tid * 4 + i * 1024;
            int kr = e >> 7, nc = e & 127;
            const float4 v = *reinterpret_cast<const float4*>(
                &B[(size_t)(k0 + kr) * N + n0 + nc]);
            Bt[(nc + 0) * GLD + kr] = (__bf16)v.x;
            Bt[(nc + 1) * GLD + kr] = (__bf16)v.y;
            Bt[(nc + 2) * GLD + kr] = (__bf16)v.z;
            Bt[(nc + 3) * GLD + kr] = (__bf16)v.w;
        }
        if (k0 + GBK < K) {  // pull next tiles toward L2/L0
            __builtin_prefetch(&A[(size_t)(row0 + (tid >> 1)) * K + k0 + GBK], 0, 0);
            __builtin_prefetch(&B[(size_t)(k0 + GBK + (tid >> 3)) * N + n0 + ((tid & 7) << 4)], 0, 0);
        }
        __syncthreads();

        v16bf af[2];
#pragma unroll
        for (int sm = 0; sm < 2; ++sm)
            af[sm] = ld_frag(&As[(wm + sm * 16) * GLD], GLD);
#pragma unroll
        for (int sn = 0; sn < 4; ++sn) {
            v16bf bf = ld_frag(&Bt[(wn + sn * 16) * GLD], GLD);
            acc[0][sn] = wmma_bf16(af[0], bf, acc[0][sn]);
            acc[1][sn] = wmma_bf16(af[1], bf, acc[1][sn]);
        }
        __syncthreads();
    }

    // ---- epilogue: C/D layout = lanes 0-15 rows 0-7, lanes 16-31 rows 8-15 ----
    const int lane = tid & 31;
    const int nn   = lane & 15;
    const int mb   = (lane >> 4) << 3;
#pragma unroll
    for (int sm = 0; sm < 2; ++sm)
#pragma unroll
        for (int sn = 0; sn < 4; ++sn)
#pragma unroll
            for (int v = 0; v < 8; ++v) {
                size_t r = (size_t)(row0 + wm + sm * 16 + mb + v);
                size_t c = (size_t)(n0 + wn + sn * 16 + nn);
                store_out(&C[r * N + c], acc[sm][sn][v]);
            }
}

// ---------------------------------------------------------------------------
// RoPE (in-place, bf16, f32 math). X is (NTOK, heads, 128); position = n % S.
// ---------------------------------------------------------------------------
__global__ __launch_bounds__(256) void rope_kernel(__bf16* __restrict__ X,
                                                   int heads, long total)
{
    long t = (long)blockIdx.x * 256 + threadIdx.x;
    if (t >= total) return;
    int  i  = (int)(t & 63);
    long nh = t >> 6;
    int  hh = (int)(nh % heads);
    long n  = nh / heads;
    int  pos = (int)(n & (SEQ - 1));

    // inv_freq = theta^(-2i/D) = exp(-(2i/128) * ln(10000))
    float inv = __expf(-((float)(2 * i) * (1.0f / 128.0f)) * 9.210340371976184f);
    float fr  = (float)pos * inv;
    float sn, cs;
    __sincosf(fr, &sn, &cs);

    __bf16* row = X + ((size_t)n * heads + hh) * HDIM;
    float x1 = (float)row[i];
    float x2 = (float)row[i + 64];
    row[i]      = (__bf16)(x1 * cs - x2 * sn);
    row[i + 64] = (__bf16)(x2 * cs + x1 * sn);
}

// ---------------------------------------------------------------------------
// Flash attention (causal, GQA 4:1).  Grid: (S/64, H, B).  Block: 128 thr =
// 4 waves; wave w owns query rows q0 = 64*bx + 16*w .. +15.  Keys processed
// in chunks of 32.
//
// K chunk staging uses the Tensor Data Mover when available: a 2D descriptor
// (tile 128 x 32 keys, data_size=2B, tensor_dim0_stride=KVD) with LDS padding
// pad_interval=5 (pad after every 64 DWORDs == one 256B row) and
// pad_amount=3 (4 DWORDs == 16B) reproduces the 136-element LDS row stride.
// Wave 0 issues the DMA; the whole block overlaps the V-transpose staging
// with it, then waits on TENSORcnt before the block barrier.
// ---------------------------------------------------------------------------
#define KLD 136     // Ks row stride (>=128, mult of 8) == 256B + 16B pad
#define VLD 40      // Vt row stride (>=32,  mult of 8)
#define PLD 40      // Pt row stride (>=32,  mult of 8)

__global__ __launch_bounds__(128) void attn_kernel(
    const __bf16* __restrict__ Q, const __bf16* __restrict__ K,
    const __bf16* __restrict__ V, __bf16* __restrict__ O)
{
    __shared__ __bf16 Ks[32 * KLD];        // [key][d]  (padded rows)
    __shared__ __bf16 Vt[128 * VLD];       // [d][key]
    __shared__ __bf16 Pt[4][16 * PLD];     // per-wave P tile [m][k]

    const int tid  = threadIdx.x;
    const int lane = tid & 31;
    const int w    = tid >> 5;
    const int b    = blockIdx.z;
    const int h    = blockIdx.y;
    const int qt   = blockIdx.x;
    const int kvh  = h >> 2;                       // GQA: head -> kv head
    const int q0   = qt * 64 + w * 16;
    const int nn   = lane & 15;
    const int mb   = (lane >> 4) << 3;
    const float scale = 0.08838834764831845f;      // 1/sqrt(128)

    // Q fragments (held in registers for the whole key loop)
    const __bf16* qbase = Q + ((size_t)b * SEQ + q0) * HID + (size_t)h * HDIM;
    v16bf qf[4];
#pragma unroll
    for (int f = 0; f < 4; ++f) qf[f] = ld_frag(qbase + f * 32, HID);

    v8f  oacc[8];
    float mrow[8], lrow[8];
#pragma unroll
    for (int d = 0; d < 8; ++d) oacc[d] = {};
#pragma unroll
    for (int v = 0; v < 8; ++v) { mrow[v] = NEG_BIG; lrow[v] = 0.0f; }

    const int nch = qt * 2 + 2;                    // chunks of 32 keys needed
    for (int ch = 0; ch < nch; ++ch) {
        const int k0 = ch * 32;
        __syncthreads();                           // LDS reuse fence

        // ---- stage K chunk (32 keys x 128d) ----
#if defined(HAVE_TDM)
        if (w == 0) {
            // D# group 0: count=1 | lds_addr | global_addr | type=2
            unsigned lds_off = (unsigned)(unsigned long long)(uintptr_t)&Ks[0];
            unsigned long long ga = (unsigned long long)(uintptr_t)(
                &K[((size_t)b * SEQ + k0) * KVD + (size_t)kvh * HDIM]);
            u32x4 g0;
            g0.x = 1u;                                   // count=1
            g0.y = lds_off;                              // lds_addr (bytes)
            g0.z = (unsigned)(ga & 0xffffffffu);         // global_addr[31:0]
            g0.w = (unsigned)((ga >> 32) & 0x01ffffffu)  // global_addr[56:32]
                 | (2u << 30);                           // type=2 (image)
            // D# group 1
            i32x8 g1;
            g1[0] = (int)((1u << 16)      // data_size = 1 (2 bytes)
                        | (1u << 20)      // pad_enable
                        | (5u << 22)      // pad_interval: after 64 DWORDs (256B)
                        | (3u << 25));    // pad_amount: 4 DWORDs (16B)
            g1[1] = (int)(128u << 16);    // tensor_dim0[15:0] = 128
            g1[2] = (int)(32u  << 16);    // tensor_dim1[15:0] = 32
            g1[3] = (int)(128u << 16);    // tile_dim0 = 128 (one 256B row)
            g1[4] = 32;                   // tile_dim1 = 32 keys (tile_dim2 = 0)
            g1[5] = KVD;                  // tensor_dim0_stride = 1024 elems
            g1[6] = 0;
            g1[7] = 0;
            i32x4 z4 = {0, 0, 0, 0};
#if defined(TDM_6ARG)
            i32x8 z8 = {0, 0, 0, 0, 0, 0, 0, 0};
            __builtin_amdgcn_tensor_load_to_lds(g0, g1, z4, z4, z8, 0);
#else
            __builtin_amdgcn_tensor_load_to_lds(g0, g1, z4, z4, 0);
#endif
        }
#else
#pragma unroll
        for (int i = 0; i < 4; ++i) {
            int e  = tid + i * 128;                // 512 v8bf total
            int ky = e >> 4;
            int dc = (e & 15) << 3;
            *reinterpret_cast<v8bf*>(&Ks[ky * KLD + dc]) =
                *reinterpret_cast<const v8bf*>(
                    &K[((size_t)b * SEQ + k0 + ky) * KVD + (size_t)kvh * HDIM + dc]);
        }
#endif
        // ---- stage V chunk transposed: Vt[d][key] (overlaps the K DMA) ----
#pragma unroll
        for (int i = 0; i < 32; ++i) {
            int e  = tid + i * 128;
            int ky = e >> 7;
            int dd = e & 127;
            Vt[dd * VLD + ky] =
                V[((size_t)b * SEQ + k0 + ky) * KVD + (size_t)kvh * HDIM + dd];
        }
#if defined(HAVE_TDM)
        if (w == 0) __builtin_amdgcn_s_wait_tensorcnt(0);
#endif
        __syncthreads();

        // ---- scores: S = Q . K^T   (two 16-key column tiles) ----
        v8f sc0 = {}, sc1 = {};
#pragma unroll
        for (int f = 0; f < 4; ++f) {
            v16bf b0 = ld_frag(&Ks[0  * KLD + f * 32], KLD);
            v16bf b1 = ld_frag(&Ks[16 * KLD + f * 32], KLD);
            sc0 = wmma_bf16(qf[f], b0, sc0);
            sc1 = wmma_bf16(qf[f], b1, sc1);
        }

        // ---- scale + causal mask ----
#pragma unroll
        for (int v = 0; v < 8; ++v) {
            int qi = q0 + mb + v;
            float s0 = sc0[v] * scale;
            float s1 = sc1[v] * scale;
            if (k0 + nn      > qi) s0 = NEG_BIG;
            if (k0 + 16 + nn > qi) s1 = NEG_BIG;
            sc0[v] = s0; sc1[v] = s1;
        }

        // ---- online softmax (row reductions across 16-lane half-groups) ----
#pragma unroll
        for (int v = 0; v < 8; ++v) {
            float mx = fmaxf(sc0[v], sc1[v]);
            mx = fmaxf(mx, __shfl_xor(mx, 1, 32));
            mx = fmaxf(mx, __shfl_xor(mx, 2, 32));
            mx = fmaxf(mx, __shfl_xor(mx, 4, 32));
            mx = fmaxf(mx, __shfl_xor(mx, 8, 32));
            float mnew  = fmaxf(mrow[v], mx);
            float alpha = __expf(mrow[v] - mnew);
            float p0 = __expf(sc0[v] - mnew);
            float p1 = __expf(sc1[v] - mnew);
            sc0[v] = p0; sc1[v] = p1;
            float rs = p0 + p1;
            rs += __shfl_xor(rs, 1, 32);
            rs += __shfl_xor(rs, 2, 32);
            rs += __shfl_xor(rs, 4, 32);
            rs += __shfl_xor(rs, 8, 32);
            lrow[v] = lrow[v] * alpha + rs;
            mrow[v] = mnew;
#pragma unroll
            for (int d = 0; d < 8; ++d) oacc[d][v] *= alpha;
        }

        // ---- C-layout -> A-layout for P via per-wave LDS tile ----
#pragma unroll
        for (int v = 0; v < 8; ++v) {
            Pt[w][(mb + v) * PLD + nn]      = (__bf16)sc0[v];
            Pt[w][(mb + v) * PLD + 16 + nn] = (__bf16)sc1[v];
        }
        // LDS ops are in-order within a wave; compiler inserts s_wait_dscnt.
        v16bf pf = ld_frag(&Pt[w][0], PLD);

        // ---- O += P . V ----
#pragma unroll
        for (int dt = 0; dt < 8; ++dt) {
            v16bf vf = ld_frag(&Vt[dt * 16 * VLD], VLD);
            oacc[dt] = wmma_bf16(pf, vf, oacc[dt]);
        }
    }

    // ---- normalize and store O (bf16, (N, H*D) layout) ----
#pragma unroll
    for (int v = 0; v < 8; ++v) {
        float inv = 1.0f / lrow[v];
        size_t row = (size_t)b * SEQ + q0 + mb + v;
#pragma unroll
        for (int dt = 0; dt < 8; ++dt)
            O[row * HID + (size_t)h * HDIM + dt * 16 + nn] =
                (__bf16)(oacc[dt][v] * inv);
    }
}

// ---------------------------------------------------------------------------
// Host launcher
// ---------------------------------------------------------------------------
extern "C" void kernel_launch(void* const* d_in, const int* in_sizes, int n_in,
                              void* d_out, int out_size, void* d_ws, size_t ws_size,
                              hipStream_t stream)
{
    (void)in_sizes; (void)n_in; (void)out_size; (void)ws_size;

    const float* hs = static_cast<const float*>(d_in[0]);
    // d_in[1] = position_ids (tiled arange; positions derived in-kernel)
    const float* wq = static_cast<const float*>(d_in[2]);
    const float* wk = static_cast<const float*>(d_in[3]);
    const float* wv = static_cast<const float*>(d_in[4]);
    const float* wo = static_cast<const float*>(d_in[5]);
    float* out = static_cast<float*>(d_out);

    char* ws = static_cast<char*>(d_ws);
    __bf16* Qb = reinterpret_cast<__bf16*>(ws);                            // 32 MB
    __bf16* Kb = reinterpret_cast<__bf16*>(ws + (size_t)32 * 1024 * 1024); //  8 MB
    __bf16* Vb = reinterpret_cast<__bf16*>(ws + (size_t)40 * 1024 * 1024); //  8 MB
    __bf16* Ob = reinterpret_cast<__bf16*>(ws + (size_t)48 * 1024 * 1024); // 32 MB

    dim3 blk256(256);

    // QKV projections (fp32 A/B -> bf16 out, bf16 WMMA, f32 accumulate)
    gemm_wmma<float, __bf16><<<dim3(HID / GBN, NTOK / GBM), blk256, 0, stream>>>(
        hs, wq, Qb, NTOK, HID, HID);
    gemm_wmma<float, __bf16><<<dim3(KVD / GBN, NTOK / GBM), blk256, 0, stream>>>(
        hs, wk, Kb, NTOK, KVD, HID);
    gemm_wmma<float, __bf16><<<dim3(KVD / GBN, NTOK / GBM), blk256, 0, stream>>>(
        hs, wv, Vb, NTOK, KVD, HID);

    // RoPE on Q and K
    {
        long totQ = (long)NTOK * HEADS * 64;
        long totK = (long)NTOK * KVH * 64;
        rope_kernel<<<(unsigned)((totQ + 255) / 256), blk256, 0, stream>>>(Qb, HEADS, totQ);
        rope_kernel<<<(unsigned)((totK + 255) / 256), blk256, 0, stream>>>(Kb, KVH, totK);
    }

    // Causal flash attention with GQA (TDM-staged K tiles when available)
    attn_kernel<<<dim3(SEQ / 64, HEADS, BATCH), dim3(128), 0, stream>>>(Qb, Kb, Vb, Ob);

    // Output projection (bf16 A, fp32 B -> fp32 out)
    gemm_wmma<__bf16, float><<<dim3(HID / GBN, NTOK / GBM), blk256, 0, stream>>>(
        Ob, wo, out, NTOK, HID, HID);
}
